// OptimizedTwoSidedVLUTLinear_29025388986484
// MI455X (gfx1250) — compile-verified, matrix operand order checked by reading, not run on hardware
//
#include <hip/hip_runtime.h>

// ---------------------------------------------------------------------------
// Two-sided vLUT linear == GEMM through the Gram matrix.
//   gram = G^T G  (8x8; folded into the weight side: 4x cheaper than B side)
//   Wg[o, i*8+e] = sum_d W[o,i,d] gram[d,e]   -> bf16, precomputed in d_ws
//   C[b,o] = sum_k A[b,k] * Wg[o,k] + bias[o] (K = 4096, bf16 WMMA, f32 acc)
// Fallback (small d_ws): gram fold fused into GEMM staging, zero ws needed.
// ---------------------------------------------------------------------------

typedef __attribute__((ext_vector_type(16))) __bf16 v16bf;
typedef __attribute__((ext_vector_type(8)))  float  v8f;

#define BDIM 4096   // batch (M)
#define ODIM 1024   // out features (N)
#define KDIM 4096   // I*D contraction
#define BM   128
#define BN   128
#define TK   32     // K per WMMA step
#define ROWU 20     // uints per padded LDS row: 64B data + 16B pad (bank spread)
#define NTHREADS 128  // 4 wave32; each wave owns a 64x64 C sub-tile (4x4 WMMA)

__device__ __forceinline__ unsigned pk2bf(float a, float b) {
  // round-to-nearest-even f32 -> bf16, packed pair
  unsigned ua = __float_as_uint(a);
  unsigned ub = __float_as_uint(b);
  ua = (ua + 0x7fffu + ((ua >> 16) & 1u)) >> 16;
  ub = (ub + 0x7fffu + ((ub >> 16) & 1u)) >> 16;
  return ua | (ub << 16);
}

union FragBF { v16bf v; uint4 q[2]; };

// ---------------------------------------------------------------------------
// Kernel 1 (fast path): Wg[o,i*8+e] = bf16( sum_d W[o,i,d] * gram[d,e] )
// One thread per (o,i) pair; gram computed per-block from G (512 FMAs).
// ---------------------------------------------------------------------------
__global__ __launch_bounds__(256)
void convw_kernel(const float* __restrict__ W,    // [1024, 512, 8]
                  const float* __restrict__ G,    // [8, 8]
                  uint4* __restrict__ Wbf)        // [1024*512] x 8 bf16
{
  __shared__ float gs[64];
  if (threadIdx.x < 64) {
    int d = threadIdx.x >> 3, e = threadIdx.x & 7;
    float s = 0.f;
#pragma unroll
    for (int k = 0; k < 8; ++k) s = __builtin_fmaf(G[k * 8 + d], G[k * 8 + e], s);
    gs[threadIdx.x] = s;
  }
  __syncthreads();

  int idx = blockIdx.x * 256 + threadIdx.x;       // o*512 + i
  const float4* p = (const float4*)W + (size_t)idx * 2;
  float4 f0 = p[0], f1 = p[1];
  float w[8] = {f0.x, f0.y, f0.z, f0.w, f1.x, f1.y, f1.z, f1.w};
  float d8[8];
#pragma unroll
  for (int e = 0; e < 8; ++e) {
    float s = 0.f;
#pragma unroll
    for (int dd = 0; dd < 8; ++dd) s = __builtin_fmaf(w[dd], gs[dd * 8 + e], s);
    d8[e] = s;
  }
  uint4 u;
  u.x = pk2bf(d8[0], d8[1]); u.y = pk2bf(d8[2], d8[3]);
  u.z = pk2bf(d8[4], d8[5]); u.w = pk2bf(d8[6], d8[7]);
  Wbf[idx] = u;
}

// ---------------------------------------------------------------------------
// Kernel 2: bf16 WMMA GEMM. 4 wave32 per WG, 128x128 WG tile, 64x64 per wave,
// double-buffered LDS, BK=32. WPRE: weights pre-folded bf16 in d_ws.
// ---------------------------------------------------------------------------
template <bool WPRE>
__global__ __launch_bounds__(NTHREADS)
void vlut_gemm(const float* __restrict__ A,      // [4096, 4096] f32
               const float* __restrict__ Wf,     // [1024, 4096] f32 (!WPRE)
               const uint4* __restrict__ Wbf,    // [1024*4096] bf16 (WPRE)
               const float* __restrict__ G,      // [8, 8]       (!WPRE)
               const float* __restrict__ bias,   // [1024]
               float* __restrict__ out)          // [4096, 1024]
{
  __shared__ unsigned ldsA[2][BM * ROWU];   // 2 x 128 rows x (64B + 16B pad)
  __shared__ unsigned ldsW[2][BN * ROWU];
  __shared__ float    gram_s[64];

  const int t     = threadIdx.x;
  const int lane  = t & 31;
  const int wave  = t >> 5;          // 0..3
  const int waveM = wave >> 1;       // 2 slabs of 64 rows
  const int waveN = wave & 1;        // 2 slabs of 64 cols
  const int halfK = lane >> 4;       // K-half select per WMMA 16-bit layout
  const int l15   = lane & 15;

  const int mBase = blockIdx.y * BM;
  const int nBase = blockIdx.x * BN;

  if constexpr (!WPRE) {
    if (t < 64) {
      int d = t >> 3, e = t & 7;
      float s = 0.f;
#pragma unroll
      for (int k = 0; k < 8; ++k) s = __builtin_fmaf(G[k * 8 + d], G[k * 8 + e], s);
      gram_s[t] = s;
    }
    __syncthreads();   // gram_s visible to fused staging below
  }

  v8f acc[4][4] = {};

  // ---- stage one 128x32 A tile (f32->bf16) + 128x32 W tile into LDS ----
  auto stage = [&](int buf, int kt) {
    unsigned* la = &ldsA[buf][0];
    unsigned* lw = &ldsW[buf][0];
#pragma unroll
    for (int it = 0; it < 4; ++it) {             // 512 chunks of 8 elems
      int c = t + it * NTHREADS;
      int row = c >> 2, ch = c & 3;
      const float4* p = (const float4*)(A + (size_t)(mBase + row) * KDIM + kt + ch * 8);
      float4 f0 = p[0], f1 = p[1];
      uint4 u;
      u.x = pk2bf(f0.x, f0.y); u.y = pk2bf(f0.z, f0.w);
      u.z = pk2bf(f1.x, f1.y); u.w = pk2bf(f1.z, f1.w);
      *((uint4*)&la[row * ROWU + ch * 4]) = u;
    }
    if constexpr (WPRE) {
#pragma unroll
      for (int it = 0; it < 4; ++it) {           // pure bf16 copy
        int c = t + it * NTHREADS;
        int row = c >> 2, ch = c & 3;
        uint4 u = Wbf[((size_t)(nBase + row) * KDIM + kt + ch * 8) >> 3];
        *((uint4*)&lw[row * ROWU + ch * 4]) = u;
      }
    } else {
#pragma unroll
      for (int it = 0; it < 4; ++it) {           // fused gram fold
        int c = t + it * NTHREADS;
        int row = c >> 2, ch = c & 3;
        const float4* p = (const float4*)(Wf + (size_t)(nBase + row) * KDIM + kt + ch * 8);
        float4 f0 = p[0], f1 = p[1];
        float w[8] = {f0.x, f0.y, f0.z, f0.w, f1.x, f1.y, f1.z, f1.w};
        float d8[8];
#pragma unroll
        for (int e = 0; e < 8; ++e) {
          float s = 0.f;
#pragma unroll
          for (int dd = 0; dd < 8; ++dd) s = __builtin_fmaf(w[dd], gram_s[dd * 8 + e], s);
          d8[e] = s;
        }
        uint4 u;
        u.x = pk2bf(d8[0], d8[1]); u.y = pk2bf(d8[2], d8[3]);
        u.z = pk2bf(d8[4], d8[5]); u.w = pk2bf(d8[6], d8[7]);
        *((uint4*)&lw[row * ROWU + ch * 4]) = u;
      }
    }
  };

  // ---- fragments per ISA 16-bit 16x32 layout + 4x4 WMMA ----
  auto compute = [&](int buf) {
    const unsigned* la = &ldsA[buf][0];
    const unsigned* lw = &ldsW[buf][0];
    FragBF afr[4], bfr[4];
#pragma unroll
    for (int mi = 0; mi < 4; ++mi) {
      const unsigned* rp = &la[(waveM * 64 + mi * 16 + l15) * ROWU];
      afr[mi].q[0] = *((const uint4*)(rp + halfK * 4));       // K 0-7 / 8-15
      afr[mi].q[1] = *((const uint4*)(rp + 8 + halfK * 4));   // K 16-23 / 24-31
    }
#pragma unroll
    for (int ni = 0; ni < 4; ++ni) {
      const unsigned* rp = &lw[(waveN * 64 + ni * 16 + l15) * ROWU];
      bfr[ni].q[0] = *((const uint4*)(rp + halfK * 4));
      bfr[ni].q[1] = *((const uint4*)(rp + 8 + halfK * 4));
    }
#pragma unroll
    for (int mi = 0; mi < 4; ++mi)
#pragma unroll
      for (int ni = 0; ni < 4; ++ni)
        acc[mi][ni] = __builtin_amdgcn_wmma_f32_16x16x32_bf16(
            false, afr[mi].v, false, bfr[ni].v, (short)0, acc[mi][ni], false, false);
  };

  // ---- main loop: ping-pong LDS, one barrier per K-step ----
  stage(0, 0);
  for (int kt = 0; kt < KDIM; kt += 2 * TK) {
    __syncthreads();
    stage(1, kt + TK);                    // always valid: KDIM % 64 == 0
    compute(0);
    __syncthreads();
    if (kt + 2 * TK < KDIM) stage(0, kt + 2 * TK);
    compute(1);
  }

  // ---- epilogue: C layout VGPR v -> M = v + 8*(lane>=16), N = lane&15 ----
#pragma unroll
  for (int ni = 0; ni < 4; ++ni) {
    int col = nBase + waveN * 64 + ni * 16 + l15;
    float bb = bias[col];
#pragma unroll
    for (int mi = 0; mi < 4; ++mi) {
      int rbase = mBase + waveM * 64 + mi * 16 + halfK * 8;
#pragma unroll
      for (int v = 0; v < 8; ++v)
        out[(size_t)(rbase + v) * ODIM + col] = acc[mi][ni][v] + bb;
    }
  }
}

// ---------------------------------------------------------------------------
extern "C" void kernel_launch(void* const* d_in, const int* in_sizes, int n_in,
                              void* d_out, int out_size, void* d_ws, size_t ws_size,
                              hipStream_t stream) {
  const float* A    = (const float*)d_in[0];  // input_encodings [4096,512,8]
  const float* W    = (const float*)d_in[1];  // weight_encodings [1024,512,8]
  const float* G    = (const float*)d_in[2];  // [8,8]
  const float* bias = (const float*)d_in[3];  // [1024]
  float* out = (float*)d_out;

  dim3 grid(ODIM / BN, BDIM / BM);            // (8, 32) = 256 workgroups

  const size_t wbf_bytes = (size_t)ODIM * KDIM * 2;   // 8 MB bf16 folded weights
  if (ws_size >= wbf_bytes) {
    uint4* Wbf = (uint4*)d_ws;
    convw_kernel<<<(ODIM * 512) / 256, 256, 0, stream>>>(W, G, Wbf);
    vlut_gemm<true><<<grid, NTHREADS, 0, stream>>>(A, W, Wbf, G, bias, out);
  } else {
    vlut_gemm<false><<<grid, NTHREADS, 0, stream>>>(A, W, nullptr, G, bias, out);
  }

  (void)in_sizes; (void)n_in; (void)out_size;
}
